// MLEI_12970801234193
// MI455X (gfx1250) — matrix operand order repaired and checked by reading, not compiled, not run on hardware
//
#include <hip/hip_runtime.h>
#include <hip/hip_bf16.h>

// ---------------------------------------------------------------------------
// CDNA5 / gfx1250 CaNet-style GNN forward pass.
// Dense math on v_wmma_f32_16x16x32_bf16; bf16 B tiles staged with
// global_load_async_to_lds_b128 (ASYNCcnt); fragments read as ds_load_b128.
// ---------------------------------------------------------------------------

typedef __attribute__((ext_vector_type(16))) __bf16 v16bf;
typedef __attribute__((ext_vector_type(8)))  __bf16 v8bf;
typedef __attribute__((ext_vector_type(8)))  float  v8f;

__device__ __forceinline__ __bf16 f2bf(float f) {
  __hip_bfloat16 h = __float2bfloat16(f);   // RNE
  __bf16 r;
  __builtin_memcpy(&r, &h, 2);
  return r;
}

__device__ __forceinline__ unsigned pk2bf(float a, float b) {
  __hip_bfloat16 x = __float2bfloat16(a), y = __float2bfloat16(b);
  unsigned short sa, sb;
  __builtin_memcpy(&sa, &x, 2);
  __builtin_memcpy(&sb, &y, 2);
  return (unsigned)sa | ((unsigned)sb << 16);   // -> v_cvt_pk_bf16_f32
}

// ======================== generic WMMA GEMM =================================
// C[M,Ncols] = act(A[M,Kd](f32, row stride lda) @ B + bias)
// B is bf16 stored TRANSPOSED: Bt[Ncols][Kd] row-major.
// block = 256 threads = 8 waves; block tile 64x64; wave tile 16x32; K-step 32.
#define TM 64
#define TN 64
#define TK 32
#define LDT 40   // LDS row stride in bf16 (80B: 16B-aligned, bank-staggered)

__global__ __launch_bounds__(256)
void wmma_gemm_bf16(const float* __restrict__ A, int lda,
                    const __bf16* __restrict__ Bt,
                    const float* __restrict__ bias,
                    float* __restrict__ C,
                    int M, int Kd, int Ncols, int relu)
{
  __shared__ __align__(16) __bf16 As[TM][LDT];   // A row-major  [m][k]
  __shared__ __align__(16) __bf16 Bs[TN][LDT];   // B transposed [n][k]

  const int bm   = blockIdx.x * TM;
  const int bn   = blockIdx.y * TN;
  const int tid  = threadIdx.x;
  const int wave = tid >> 5;
  const int lane = tid & 31;
  const int lm   = lane & 15;
  const int lh   = lane >> 4;
  const int wm   = (wave & 3) * 16;
  const int wn   = (wave >> 2) * 32;

  // staging maps. Out-of-range rows/cols are CLAMPED, not branched: they feed
  // only output rows/cols that are never stored, so garbage there is harmless.
  const int ar0 = tid >> 3;                         // A row 0..31 (+32)
  const int ac4 = (tid & 7) * 4;                    // A float4 col group
  const int am0 = min(bm + ar0,      M - 1);
  const int am1 = min(bm + ar0 + 32, M - 1);
  const int bnc = min(bn + (tid >> 2), Ncols - 1);  // B row (n), clamped
  const int bch = (tid & 3) * 8;                    // B 8-elem k-chunk

  const __bf16* bsrc = Bt + (long)bnc * Kd + bch;   // global src (16B aligned)
  const unsigned blds = (unsigned)(size_t)(&Bs[tid >> 2][bch]);  // LDS offset

  v8f acc0 = {};
  v8f acc1 = {};

  for (int k0 = 0; k0 < Kd; k0 += TK) {
    // ---- B tile: pure bf16 copy -> async global->LDS (no VGPR round trip) --
    {
      const __bf16* g = bsrc + k0;
      asm volatile("global_load_async_to_lds_b128 %0, %1, off"
                   :: "v"(blds), "v"(g) : "memory");
    }
    // ---- A tile: 64x32 f32 -> bf16, float4 loads + b64 LDS stores ----------
    {
      const float* a0 = A + (long)am0 * lda + (k0 + ac4);
      const float* a1 = A + (long)am1 * lda + (k0 + ac4);
      if (k0 + TK < Kd) {
        __builtin_prefetch(a0 + TK, 0, 0);   // global_prefetch_b8
        __builtin_prefetch(a1 + TK, 0, 0);
      }
      float4 f0 = *(const float4*)a0;
      float4 f1 = *(const float4*)a1;
      uint2 p0, p1;
      p0.x = pk2bf(f0.x, f0.y); p0.y = pk2bf(f0.z, f0.w);
      p1.x = pk2bf(f1.x, f1.y); p1.y = pk2bf(f1.z, f1.w);
      *(uint2*)(&As[ar0][ac4])      = p0;
      *(uint2*)(&As[ar0 + 32][ac4]) = p1;
    }
    asm volatile("s_wait_asynccnt 0x0" ::: "memory");  // own async done
    __syncthreads();                                   // => all waves' done

    // ---- fragments: contiguous 16B runs per §7.12.2 ----
    union { v16bf v; v8bf h[2]; } ua, ub0, ub1;
    const v8bf* arow = (const v8bf*)(&As[wm + lm][0]);
    ua.h[0] = arow[lh];        // K = 8*lh   .. +7
    ua.h[1] = arow[lh + 2];    // K = 16+8*lh .. +7
    const v8bf* br0 = (const v8bf*)(&Bs[wn + lm][0]);
    ub0.h[0] = br0[lh];        // K = 8*lh   .. +7
    ub0.h[1] = br0[lh + 2];    // K = 16+8*lh .. +7
    const v8bf* br1 = (const v8bf*)(&Bs[wn + 16 + lm][0]);
    ub1.h[0] = br1[lh];
    ub1.h[1] = br1[lh + 2];

    acc0 = __builtin_amdgcn_wmma_f32_16x16x32_bf16(false, ua.v, false, ub0.v,
                                                   (short)0, acc0, false, false);
    acc1 = __builtin_amdgcn_wmma_f32_16x16x32_bf16(false, ua.v, false, ub1.v,
                                                   (short)0, acc1, false, false);
    __syncthreads();
  }

  #pragma unroll
  for (int r = 0; r < 8; ++r) {
    int gm = bm + wm + r + 8 * lh;
    if (gm >= M) continue;
    int gn0 = bn + wn + lm;
    int gn1 = gn0 + 16;
    if (gn0 < Ncols) {
      float o = acc0[r] + (bias ? bias[gn0] : 0.0f);
      C[(long)gm * Ncols + gn0] = relu ? fmaxf(o, 0.0f) : o;
    }
    if (gn1 < Ncols) {
      float o = acc1[r] + (bias ? bias[gn1] : 0.0f);
      C[(long)gm * Ncols + gn1] = relu ? fmaxf(o, 0.0f) : o;
    }
  }
}

// ================= kTv = k^T @ v  (split-K WMMA, 128x128 out) ===============
// qkv layout [N][384]: q @ +0, k @ +128, v @ +256.
// Both tiles staged feature-major so A and B fragments are contiguous.
// K-dim padding must be ZERO here (enters the reduction), so guards stay.
__global__ __launch_bounds__(256)
void ktv_wmma(const float* __restrict__ qkv, float* __restrict__ ktv, int Nrows)
{
  __shared__ __align__(16) __bf16 Ks[128][LDT];  // Ks[m][kk] = k[r0+kk][m]
  __shared__ __align__(16) __bf16 Vs[128][LDT];  // Vs[n][kk] = v[r0+kk][n]

  const int tid  = threadIdx.x;
  const int wave = tid >> 5;
  const int lane = tid & 31;
  const int lm   = lane & 15;
  const int lh   = lane >> 4;

  v8f acc[8];
  #pragma unroll
  for (int t = 0; t < 8; ++t) acc[t] = (v8f){};

  const int chunks = (Nrows + 31) / 32;
  for (int ch = blockIdx.x; ch < chunks; ch += gridDim.x) {
    const long r0 = (long)ch * 32;
    #pragma unroll
    for (int i = tid; i < 32 * 128; i += 256) {
      int r = i >> 7, c = i & 127;             // coalesced over c
      float kv = 0.0f, vv = 0.0f;
      if (r0 + r < Nrows) {
        kv = qkv[(r0 + r) * 384 + 128 + c];
        vv = qkv[(r0 + r) * 384 + 256 + c];
      }
      Ks[c][r] = f2bf(kv);
      Vs[c][r] = f2bf(vv);
    }
    __syncthreads();

    union { v16bf v; v8bf h[2]; } ua;
    const v8bf* arow = (const v8bf*)(&Ks[16 * wave + lm][0]);
    ua.h[0] = arow[lh];
    ua.h[1] = arow[lh + 2];
    #pragma unroll
    for (int t = 0; t < 8; ++t) {
      union { v16bf v; v8bf h[2]; } ub;
      const v8bf* brow = (const v8bf*)(&Vs[16 * t + lm][0]);
      ub.h[0] = brow[lh];
      ub.h[1] = brow[lh + 2];
      acc[t] = __builtin_amdgcn_wmma_f32_16x16x32_bf16(false, ua.v, false, ub.v,
                                                       (short)0, acc[t], false, false);
    }
    __syncthreads();
  }

  #pragma unroll
  for (int t = 0; t < 8; ++t)
    #pragma unroll
    for (int r = 0; r < 8; ++r) {
      int m = 16 * wave + r + 8 * lh;
      int n = 16 * t + lm;
      atomicAdd(&ktv[m * 128 + n], acc[t][r]);
    }
}

// ============================ graph kernels =================================
__global__ void deg_count(const int* __restrict__ col, float* __restrict__ deg, int E_) {
  int e = blockIdx.x * blockDim.x + threadIdx.x;
  if (e < E_) atomicAdd(&deg[col[e]], 1.0f);
}

__global__ void deg_inv(const float* __restrict__ deg, float* __restrict__ inv, int n) {
  int i = blockIdx.x * blockDim.x + threadIdx.x;
  if (i < n) { float d = deg[i]; inv[i] = d > 0.0f ? rsqrtf(d) : 0.0f; }
}

// one edge per 32 lanes, float4 per lane (H=128)
__global__ void gcn_scatter(const float* __restrict__ h, const int* __restrict__ row,
                            const int* __restrict__ col, const float* __restrict__ inv,
                            float* __restrict__ out, int E_) {
  long idx  = (long)blockIdx.x * blockDim.x + threadIdx.x;
  long edge = idx >> 5;
  int  lane = (int)(idx & 31);
  if (edge >= E_) return;
  int r = row[edge], c = col[edge];
  float val = inv[r] * inv[c];
  const float4 hv = *(const float4*)(h + (long)r * 128 + lane * 4);
  float* dst = out + (long)c * 128 + lane * 4;
  atomicAdd(dst + 0, hv.x * val);
  atomicAdd(dst + 1, hv.y * val);
  atomicAdd(dst + 2, hv.z * val);
  atomicAdd(dst + 3, hv.w * val);
}

__global__ void concat2(const float* __restrict__ a, const float* __restrict__ b,
                        float* __restrict__ o, int n) {
  long i = (long)blockIdx.x * blockDim.x + threadIdx.x;
  long tot = (long)n * 128;
  if (i >= tot) return;
  long r = i >> 7; int c = (int)(i & 127);
  o[(r << 8) + c]       = a[i];
  o[(r << 8) + 128 + c] = b[i];
}

// out[n][j] = act( resid + sum_k env[n][k] * big[n][k*128+j] ), K=4
__global__ void combine_env(const float* __restrict__ big, const float* __restrict__ env,
                            const float* __restrict__ resid, float* __restrict__ out,
                            int n, int relu) {
  long i = (long)blockIdx.x * blockDim.x + threadIdx.x;
  long tot = (long)n * 128;
  if (i >= tot) return;
  long node = i >> 7; int j = (int)(i & 127);
  const float* e = env + node * 4;
  const float* g = big + (node << 9);
  float s = resid ? resid[i] : 0.0f;
  s += e[0] * g[j] + e[1] * g[128 + j] + e[2] * g[256 + j] + e[3] * g[384 + j];
  out[i] = relu ? fmaxf(s, 0.0f) : s;
}

// logits = h @ w[H,4] + b, softmax over K=4 (H=128)
__global__ void env_softmax(const float* __restrict__ h, const float* __restrict__ w,
                            const float* __restrict__ b, float* __restrict__ env, int n) {
  int node = blockIdx.x * blockDim.x + threadIdx.x;
  if (node >= n) return;
  float l0 = b[0], l1 = b[1], l2 = b[2], l3 = b[3];
  const float* hr = h + (long)node * 128;
  #pragma unroll 4
  for (int j = 0; j < 128; ++j) {
    float hv = hr[j];
    const float* wr = w + j * 4;
    l0 += hv * wr[0]; l1 += hv * wr[1]; l2 += hv * wr[2]; l3 += hv * wr[3];
  }
  float m = fmaxf(fmaxf(l0, l1), fmaxf(l2, l3));
  float e0 = __expf(l0 - m), e1 = __expf(l1 - m), e2 = __expf(l2 - m), e3 = __expf(l3 - m);
  float s = e0 + e1 + e2 + e3;
  float* er = env + (long)node * 4;
  er[0] = e0 / s; er[1] = e1 / s; er[2] = e2 / s; er[3] = e3 / s;
}

// ====================== linear-transformer helpers ==========================
__global__ void sumsq_block(const float* __restrict__ x, int n, int cols, int ld,
                            float* __restrict__ out) {
  float s = 0.0f;
  long total = (long)n * cols;
  for (long i = (long)blockIdx.x * blockDim.x + threadIdx.x; i < total;
       i += (long)gridDim.x * blockDim.x) {
    long r = i / cols; int c = (int)(i - r * cols);
    float v = x[r * ld + c];
    s += v * v;
  }
  for (int off = 16; off > 0; off >>= 1) s += __shfl_down(s, off, 32);
  __shared__ float red[8];
  int lane = threadIdx.x & 31, wave = threadIdx.x >> 5;
  if (lane == 0) red[wave] = s;
  __syncthreads();
  if (wave == 0) {
    s = (lane < (int)(blockDim.x >> 5)) ? red[lane] : 0.0f;
    for (int off = 4; off > 0; off >>= 1) s += __shfl_down(s, off, 32);
    if (lane == 0) atomicAdd(out, s);
  }
}

__global__ void compute_scales(float* __restrict__ scal) {
  if (threadIdx.x == 0) {
    scal[2] = 1.0f / fmaxf(sqrtf(scal[0]), 1e-12f);  // q scale
    scal[3] = 1.0f / fmaxf(sqrtf(scal[1]), 1e-12f);  // k scale
  }
}

__global__ void colsum128(const float* __restrict__ x, int n, int ld, int col0,
                          float* __restrict__ out) {
  int t = threadIdx.x;
  long r0 = (long)blockIdx.x * 256;
  float s = 0.0f;
  for (int r = 0; r < 256; ++r) {
    long rr = r0 + r;
    if (rr < n) s += x[rr * ld + col0 + t];
  }
  atomicAdd(&out[t], s);
}

__global__ void dot_qksum(const float* __restrict__ qkv, const float* __restrict__ ksum,
                          float* __restrict__ dotq, int n) {
  int node = blockIdx.x * blockDim.x + threadIdx.x;
  if (node >= n) return;
  const float* qr = qkv + (long)node * 384;
  float s = 0.0f;
  #pragma unroll 4
  for (int j = 0; j < 128; ++j) s += qr[j] * ksum[j];
  dotq[node] = s;
}

// gr = (v + c1*T) / max(1 + c1*dotq, eps), c1 = qs*ks/n
__global__ void compute_gr(const float* __restrict__ qkv, const float* __restrict__ T,
                           const float* __restrict__ dotq, const float* __restrict__ scal,
                           float* __restrict__ gr, int n) {
  long i = (long)blockIdx.x * blockDim.x + threadIdx.x;
  long tot = (long)n * 128;
  if (i >= tot) return;
  long node = i >> 7; int j = (int)(i & 127);
  float c1 = scal[2] * scal[3] / (float)n;
  float num = qkv[node * 384 + 256 + j] + c1 * T[i];
  float den = fmaxf(1.0f + c1 * dotq[node], 1e-12f);
  gr[i] = num / den;
}

// ====================== weight packing (transposed bf16) ====================
// dst[Ncols][Kd] <- src[Kd][Ncols]
__global__ void pack_bT(const float* __restrict__ src, __bf16* __restrict__ dst,
                        int Kd, int Ncols) {
  long i = (long)blockIdx.x * blockDim.x + threadIdx.x;
  long tot = (long)Kd * Ncols;
  if (i >= tot) return;
  long r = i / Ncols; int c = (int)(i % Ncols);
  dst[(long)c * Kd + r] = f2bf(src[i]);
}

// src [L][K][D][H] -> dst[l] = Bt[(k*H+h)][d]
__global__ void pack_conv_T(const float* __restrict__ src, __bf16* __restrict__ dst,
                            int Lc, int Kc, int Dc, int Hc) {
  long idx = (long)blockIdx.x * blockDim.x + threadIdx.x;
  long tot = (long)Lc * Kc * Dc * Hc;
  if (idx >= tot) return;
  int hcol = (int)(idx % Hc);
  long t = idx / Hc;
  int d = (int)(t % Dc); t /= Dc;
  int k = (int)(t % Kc); t /= Kc;
  int l = (int)t;
  dst[(long)l * Kc * Hc * Dc + ((long)k * Hc + hcol) * Dc + d] = f2bf(src[idx]);
}

// qkv: dst[3H][H], rows [0,H) = q^T, [H,2H) = k^T, [2H,3H) = v^T
__global__ void pack_qkv_T(const float* __restrict__ q, const float* __restrict__ k,
                           const float* __restrict__ v, __bf16* __restrict__ dst, int H_) {
  long i = (long)blockIdx.x * blockDim.x + threadIdx.x;
  long tot = (long)H_ * H_;
  if (i >= tot) return;
  long r = i / H_; int c = (int)(i % H_);
  dst[(long)c * H_ + r]               = f2bf(q[i]);
  dst[((long)H_ + c) * H_ + r]        = f2bf(k[i]);
  dst[((long)2 * H_ + c) * H_ + r]    = f2bf(v[i]);
}

__global__ void concat3_f32(const float* __restrict__ a, const float* __restrict__ b,
                            const float* __restrict__ c, float* __restrict__ o, int m) {
  int i = blockIdx.x * blockDim.x + threadIdx.x;
  if (i < m) { o[i] = a[i]; o[m + i] = b[i]; o[2 * m + i] = c[i]; }
}

__global__ void avg2(const float* __restrict__ a, const float* __restrict__ b,
                     float* __restrict__ o, long cnt) {
  long i = (long)blockIdx.x * blockDim.x + threadIdx.x;
  if (i < cnt) o[i] = 0.5f * (a[i] + b[i]);
}

// ================================ host side =================================
extern "C" void kernel_launch(void* const* d_in, const int* in_sizes, int n_in,
                              void* d_out, int out_size, void* d_ws, size_t ws_size,
                              hipStream_t stream) {
  const float* x      = (const float*)d_in[0];
  const int*   ei     = (const int*)  d_in[1];
  const float* fc0_w  = (const float*)d_in[2];
  const float* fc0_b  = (const float*)d_in[3];
  const float* fc1_w  = (const float*)d_in[4];
  const float* fc1_b  = (const float*)d_in[5];
  const float* env_w  = (const float*)d_in[6];
  const float* env_b  = (const float*)d_in[7];
  const float* conv_w = (const float*)d_in[8];
  const float* q_w    = (const float*)d_in[9];
  const float* q_b    = (const float*)d_in[10];
  const float* k_w    = (const float*)d_in[11];
  const float* k_b    = (const float*)d_in[12];
  const float* v_w    = (const float*)d_in[13];
  const float* v_b    = (const float*)d_in[14];
  const float* envp_w = (const float*)d_in[15];
  const float* envp_b = (const float*)d_in[16];
  const float* genv_w = (const float*)d_in[17];
  const float* gt_w   = (const float*)d_in[18];
  const float* gt_b   = (const float*)d_in[19];

  const int H = in_sizes[3];             // 128
  const int D = in_sizes[2] / H;         // 256
  const int C = in_sizes[5];             // 40
  const int K = in_sizes[16];            // 4
  const int L = in_sizes[7] / K;         // 2
  const int N = in_sizes[0] / D;         // 50000
  const int E = in_sizes[1] / 2;         // 800000
  (void)n_in; (void)ws_size; (void)out_size;

  const int* row = ei;
  const int* col = ei + E;

  char* base = (char*)d_ws;
  size_t off = 0;
  auto alloc = [&](size_t bytes) -> void* {
    off = (off + 255) & ~(size_t)255;
    void* p = base + off;
    off += bytes;
    return p;
  };
  __bf16* bf_fc0  = (__bf16*)alloc((size_t)D * H * 2);             // [H][D]
  __bf16* bf_conv = (__bf16*)alloc((size_t)L * 2 * H * K * H * 2); // [l][K*H][2H]
  __bf16* bf_qkv  = (__bf16*)alloc((size_t)H * 3 * H * 2);         // [3H][H]
  __bf16* bf_genv = (__bf16*)alloc((size_t)H * K * H * 2);         // [K*H][H]
  __bf16* bf_gt   = (__bf16*)alloc((size_t)H * H * 2);             // [H][H]
  __bf16* bf_fc1  = (__bf16*)alloc((size_t)H * C * 2);             // [C][H]
  __bf16* bf_ktv  = (__bf16*)alloc((size_t)H * H * 2);             // [H][H] (B^T)
  float*  qkv_b   = (float*)alloc((size_t)3 * H * 4);
  float*  f_h     = (float*)alloc((size_t)N * H * 4);
  float*  f_gcn   = (float*)alloc((size_t)N * H * 4);              // reused: T, ctx
  float*  f_hi    = (float*)alloc((size_t)N * 2 * H * 4);          // reused: gr | gh
  float*  f_big   = (float*)alloc((size_t)N * K * H * 4);
  float*  f_qkv   = (float*)alloc((size_t)N * 3 * H * 4);
  float*  f_env   = (float*)alloc((size_t)N * K * 4);
  float*  f_deg   = (float*)alloc((size_t)N * 4);
  float*  f_inv   = (float*)alloc((size_t)N * 4);
  float*  f_dotq  = (float*)alloc((size_t)N * 4);
  float*  f_ksum  = (float*)alloc((size_t)H * 4);
  float*  f_ktv   = (float*)alloc((size_t)H * H * 4);
  float*  f_scal  = (float*)alloc(4 * 4);
  float*  f_loc   = (float*)alloc((size_t)N * C * 4);
  float*  f_glob  = (float*)alloc((size_t)N * C * 4);

  auto cdiv = [](long a, long b) { return (int)((a + b - 1) / b); };
  const dim3 blk(256);

  // --- 1. weight packing (transposed bf16) ---
  pack_bT<<<cdiv((long)D * H, 256), blk, 0, stream>>>(fc0_w, bf_fc0, D, H);
  pack_conv_T<<<cdiv((long)L * K * 2 * H * H, 256), blk, 0, stream>>>(
      conv_w, bf_conv, L, K, 2 * H, H);
  pack_qkv_T<<<cdiv((long)H * H, 256), blk, 0, stream>>>(q_w, k_w, v_w, bf_qkv, H);
  pack_conv_T<<<cdiv((long)K * H * H, 256), blk, 0, stream>>>(genv_w, bf_genv, 1, K, H, H);
  pack_bT<<<cdiv((long)H * H, 256), blk, 0, stream>>>(gt_w, bf_gt, H, H);
  pack_bT<<<cdiv((long)H * C, 256), blk, 0, stream>>>(fc1_w, bf_fc1, H, C);
  concat3_f32<<<cdiv(H, 128), dim3(128), 0, stream>>>(q_b, k_b, v_b, qkv_b, H);

  // --- 2. degree / normalization ---
  hipMemsetAsync(f_deg, 0, (size_t)N * 4, stream);
  deg_count<<<cdiv(E, 256), blk, 0, stream>>>(col, f_deg, E);
  deg_inv<<<cdiv(N, 256), blk, 0, stream>>>(f_deg, f_inv, N);

  // --- 3. h = relu(x @ fc0 + b) ---
  wmma_gemm_bf16<<<dim3(cdiv(N, TM), cdiv(H, TN)), blk, 0, stream>>>(
      x, D, bf_fc0, fc0_b, f_h, N, D, H, 1);

  // --- 4. L graph-conv layers ---
  for (int i = 0; i < L; ++i) {
    env_softmax<<<cdiv(N, 256), blk, 0, stream>>>(f_h, env_w + (long)i * H * K,
                                                  env_b + (long)i * K, f_env, N);
    hipMemsetAsync(f_gcn, 0, (size_t)N * H * 4, stream);
    gcn_scatter<<<cdiv((long)E * 32, 256), blk, 0, stream>>>(f_h, row, col, f_inv, f_gcn, E);
    concat2<<<cdiv((long)N * H, 256), blk, 0, stream>>>(f_gcn, f_h, f_hi, N);
    wmma_gemm_bf16<<<dim3(cdiv(N, TM), cdiv(K * H, TN)), blk, 0, stream>>>(
        f_hi, 2 * H, bf_conv + (size_t)i * 2 * H * K * H, nullptr, f_big, N, 2 * H, K * H, 0);
    combine_env<<<cdiv((long)N * H, 256), blk, 0, stream>>>(f_big, f_env, f_h, f_h, N, 1);
  }

  // --- 5. local logits ---
  wmma_gemm_bf16<<<dim3(cdiv(N, TM), cdiv(C, TN)), blk, 0, stream>>>(
      f_h, H, bf_fc1, fc1_b, f_loc, N, H, C, 0);

  // --- 6. linear transformer ---
  wmma_gemm_bf16<<<dim3(cdiv(N, TM), cdiv(3 * H, TN)), blk, 0, stream>>>(
      f_h, H, bf_qkv, qkv_b, f_qkv, N, H, 3 * H, 0);

  hipMemsetAsync(f_scal, 0, 16, stream);
  sumsq_block<<<256, blk, 0, stream>>>(f_qkv + 0, N, H, 3 * H, &f_scal[0]);
  sumsq_block<<<256, blk, 0, stream>>>(f_qkv + H, N, H, 3 * H, &f_scal[1]);
  compute_scales<<<1, dim3(32), 0, stream>>>(f_scal);

  hipMemsetAsync(f_ksum, 0, (size_t)H * 4, stream);
  colsum128<<<cdiv(N, 256), dim3(128), 0, stream>>>(f_qkv, N, 3 * H, H, f_ksum);
  dot_qksum<<<cdiv(N, 256), blk, 0, stream>>>(f_qkv, f_ksum, f_dotq, N);

  hipMemsetAsync(f_ktv, 0, (size_t)H * H * 4, stream);
  ktv_wmma<<<64, blk, 0, stream>>>(f_qkv, f_ktv, N);
  pack_bT<<<cdiv((long)H * H, 256), blk, 0, stream>>>(f_ktv, bf_ktv, H, H);

  // T = q_raw @ kTv_raw   (scales folded in later)
  wmma_gemm_bf16<<<dim3(cdiv(N, TM), cdiv(H, TN)), blk, 0, stream>>>(
      f_qkv, 3 * H, bf_ktv, nullptr, f_gcn, N, H, H, 0);

  float* f_gr = f_hi;
  float* f_gh = f_hi + (size_t)N * H;
  compute_gr<<<cdiv((long)N * H, 256), blk, 0, stream>>>(f_qkv, f_gcn, f_dotq, f_scal, f_gr, N);

  // --- 7. global env mixture ---
  env_softmax<<<cdiv(N, 256), blk, 0, stream>>>(f_gr, envp_w, envp_b, f_env, N);
  wmma_gemm_bf16<<<dim3(cdiv(N, TM), cdiv(K * H, TN)), blk, 0, stream>>>(
      f_gr, H, bf_genv, nullptr, f_big, N, H, K * H, 0);
  combine_env<<<cdiv((long)N * H, 256), blk, 0, stream>>>(f_big, f_env, nullptr, f_gcn, N, 0);

  // --- 8. global head ---
  wmma_gemm_bf16<<<dim3(cdiv(N, TM), cdiv(H, TN)), blk, 0, stream>>>(
      f_gcn, H, bf_gt, gt_b, f_gh, N, H, H, 1);
  wmma_gemm_bf16<<<dim3(cdiv(N, TM), cdiv(C, TN)), blk, 0, stream>>>(
      f_gh, H, bf_fc1, fc1_b, f_glob, N, H, C, 0);

  // --- 9. fused prediction ---
  avg2<<<cdiv((long)N * C, 256), blk, 0, stream>>>(f_loc, f_glob, (float*)d_out, (long)N * C);
}
